// MultiHeadTransformerBlock_77352361001562
// MI455X (gfx1250) — matrix operand order repaired and checked by reading, not compiled, not run on hardware
//
#include <hip/hip_runtime.h>
#include <stdint.h>

typedef __attribute__((ext_vector_type(16))) _Float16 v16h;
typedef __attribute__((ext_vector_type(8)))  float    v8f;
typedef __attribute__((ext_vector_type(4)))  unsigned int v4u;
typedef __attribute__((ext_vector_type(8)))  int      v8i;
typedef __attribute__((ext_vector_type(4)))  int      v4i;

#define B_      4
#define N_      8192
#define DIM_    128
#define DOUT_   256
#define H_      4
#define HD_     64
#define KNN_    16
#define M_      (B_ * N_)

union FragH { v16h v; uint32_t u[8]; };

// ---------------------------------------------------------------------------
// TDM: 2D f16 tile load global->LDS, row pad of 1 dword every 16 dwords
// (matches LDS row stride of BK+2 halfs). amdgpu-toolchain 6-arg builtin form.
// ---------------------------------------------------------------------------
static __device__ __forceinline__ void tdm_load_tile_f16(
    uint32_t lds_off, const _Float16* gptr,
    uint32_t tensor_d0, uint32_t tensor_d1,
    uint32_t tile_d0, uint32_t tile_d1, uint32_t stride_elems)
{
  const uint64_t ga = (uint64_t)(size_t)gptr;
  v4u g0;
  g0.x = 1u;                                   // count=1, user-mode, no gather
  g0.y = lds_off;                              // lds_addr (bytes)
  g0.z = (uint32_t)ga;                         // global_addr[31:0]
  g0.w = (uint32_t)(ga >> 32) | (2u << 30);    // global_addr[56:32] | type=2
  v8i g1;
  g1[0] = (int)((1u << 16)                     // data_size = 2 bytes
              | (1u << 20)                     // pad_enable
              | (3u << 22)                     // pad_interval: 16 dwords (64B row)
              | (0u << 25));                   // pad_amount: 1 dword
  g1[1] = (int)((tensor_d0 & 0xFFFFu) << 16);                       // dim0 lo16
  g1[2] = (int)((tensor_d0 >> 16) | ((tensor_d1 & 0xFFFFu) << 16)); // dim0 hi|dim1 lo
  g1[3] = (int)((tensor_d1 >> 16) | (tile_d0 << 16));               // dim1 hi|tile0
  g1[4] = (int)(tile_d1 & 0xFFFFu);                                 // tile1, tile2=0
  g1[5] = (int)stride_elems;                                        // dim0_stride lo32
  g1[6] = 0;
  g1[7] = 0;
  const v4i z4 = {0, 0, 0, 0};
  const v8i z8 = {0, 0, 0, 0, 0, 0, 0, 0};
  __builtin_amdgcn_tensor_load_to_lds(g0, g1, z4, z4, z8, 0);
}

// ---------------------------------------------------------------------------
// KNN: per-thread query, LDS-tiled candidate scan, register sorted top-16
// ---------------------------------------------------------------------------
#define QT 128
#define CT 256
__global__ __launch_bounds__(QT) void knn_kernel(const float* __restrict__ xyz,
                                                 int* __restrict__ knn_out)
{
  const int b = blockIdx.y;
  const int q = blockIdx.x * QT + threadIdx.x;
  const float* base = xyz + (size_t)b * N_ * 3;
  const float qx = base[q*3+0], qy = base[q*3+1], qz = base[q*3+2];

  float dist[KNN_];
  int   idx[KNN_];
#pragma unroll
  for (int i = 0; i < KNN_; ++i) { dist[i] = 3.0e38f; idx[i] = 0; }

  __shared__ float sx[CT * 3];
  for (int c0 = 0; c0 < N_; c0 += CT) {
    for (int t = threadIdx.x; t < CT*3; t += QT) sx[t] = base[(size_t)c0*3 + t];
    __syncthreads();
    for (int c = 0; c < CT; ++c) {
      const float dx = sx[c*3+0]-qx, dy = sx[c*3+1]-qy, dz = sx[c*3+2]-qz;
      const float d  = dx*dx + dy*dy + dz*dz;
      if (d < dist[KNN_-1]) {            // strict < keeps stable tie order
        dist[KNN_-1] = d; idx[KNN_-1] = c0 + c;
#pragma unroll
        for (int j = KNN_-1; j >= 1; --j) {   // single bubble pass inserts
          if (dist[j] < dist[j-1]) {
            float td = dist[j]; dist[j] = dist[j-1]; dist[j-1] = td;
            int   ti = idx[j];  idx[j]  = idx[j-1];  idx[j-1]  = ti;
          }
        }
      }
    }
    __syncthreads();
  }
#pragma unroll
  for (int i = 0; i < KNN_; ++i)
    knn_out[((size_t)b*N_ + q)*KNN_ + i] = idx[i];
}

// ---------------------------------------------------------------------------
// One-shot f32 -> f16 converter (packed 4-wide)
// ---------------------------------------------------------------------------
__global__ __launch_bounds__(256) void cvt_f32_f16_kernel(
    const float* __restrict__ x, _Float16* __restrict__ y)
{
  const int i = blockIdx.x * 256 + threadIdx.x;    // index of float4 group
  const float4 v = ((const float4*)x)[i];
  union { _Float16 h[4]; uint64_t u; } p;
  p.h[0] = (_Float16)v.x; p.h[1] = (_Float16)v.y;
  p.h[2] = (_Float16)v.z; p.h[3] = (_Float16)v.w;
  ((uint64_t*)y)[i] = p.u;
}

// ---------------------------------------------------------------------------
// GEMM: C = act(A_h[MxK](f16) @ W[KxN](f32) + bias), f32 accum via WMMA.
// A tile staged by the Tensor Data Mover; W tile staged f32->f16 (float4).
// block = 256 threads (8 waves), tile 128x64, K-step 32.
// ---------------------------------------------------------------------------
#define BM 128
#define BN 64
#define BK 32
__global__ __launch_bounds__(256) void gemm_wmma_kernel(
    const _Float16* __restrict__ A, const float* __restrict__ W,
    const float* __restrict__ bias, float* __restrict__ C,
    _Float16* __restrict__ C16, int M, int K, int N, int relu)
{
  __shared__ _Float16 As[BM][BK + 2];   // row-major (m,k); TDM pads 1 dword/row
  __shared__ _Float16 Wt[BN][BK + 2];   // transposed (n,k)

  const int tid  = threadIdx.x;
  const int lane = tid & 31;
  const int wave = tid >> 5;
  const int hi   = lane >> 4;
  const int l16  = lane & 15;
  const int bm   = blockIdx.x * BM;
  const int bn   = blockIdx.y * BN;

  const v8f vzero = {0.f,0.f,0.f,0.f,0.f,0.f,0.f,0.f};
  v8f acc[4] = {vzero, vzero, vzero, vzero};

  const uint32_t lds_as = (uint32_t)(size_t)(&As[0][0]);

  for (int k0 = 0; k0 < K; k0 += BK) {
    // --- TDM: async DMA of the 128x32 f16 A tile into padded LDS rows ---
    if (wave == 0) {
      tdm_load_tile_f16(lds_as, &A[(size_t)bm * K + k0],
                        (uint32_t)K, (uint32_t)M, BK, BM, (uint32_t)K);
    }
    // --- stage W tile (f32 -> f16, transposed) while the TDM runs ---
    for (int t = tid; t < (BK*BN)/4; t += 256) {
      const int r  = t >> 4;          // k row (16 float4 per 64-wide row)
      const int c4 = (t & 15) * 4;    // n col
      const float4 w4 = *(const float4*)&W[(size_t)(k0 + r) * N + (bn + c4)];
      Wt[c4+0][r] = (_Float16)w4.x; Wt[c4+1][r] = (_Float16)w4.y;
      Wt[c4+2][r] = (_Float16)w4.z; Wt[c4+3][r] = (_Float16)w4.w;
    }
    if (wave == 0) __builtin_amdgcn_s_wait_tensorcnt(0);
    __syncthreads();

    // A fragment (16x32 f16): lane m = l16, dword j -> K = 16*hi + 2*j
    FragH a;
#pragma unroll
    for (int j = 0; j < 8; ++j) {
      const int kk = 16*hi + 2*j;
      a.u[j] = *(const uint32_t*)&As[wave*16 + l16][kk];
    }
#pragma unroll
    for (int nt = 0; nt < 4; ++nt) {
      // B fragment (32x16 f16): lane n = l16, dword j -> K = 16*hi + 2*j
      FragH bf;
#pragma unroll
      for (int j = 0; j < 8; ++j) {
        const int kk = 16*hi + 2*j;
        bf.u[j] = *(const uint32_t*)&Wt[nt*16 + l16][kk];
      }
      acc[nt] = __builtin_amdgcn_wmma_f32_16x16x32_f16(
          false, a.v, false, bf.v, (short)0, acc[nt], false, false);
    }
    __syncthreads();
  }

  // C/D layout: VGPR r, lane -> m = r + 8*hi, n = l16
#pragma unroll
  for (int nt = 0; nt < 4; ++nt) {
    const int ncol = bn + nt*16 + l16;
    const float bb = bias[ncol];
#pragma unroll
    for (int r = 0; r < 8; ++r) {
      const int mrow = bm + wave*16 + r + 8*hi;
      float v = acc[nt][r] + bb;
      if (relu) v = fmaxf(v, 0.f);
      if (C16) C16[(size_t)mrow * N + ncol] = (_Float16)v;
      else     C  [(size_t)mrow * N + ncol] = v;
    }
  }
}

// ---------------------------------------------------------------------------
// Attention: one wave per point. Pos-MLP layer2 (16x64 @ 64x64) via WMMA.
// Emits attn_feat in f16 (feeds the FFN GEMM) and attn_out (B,H,N,K) f32.
// ---------------------------------------------------------------------------
__global__ __launch_bounds__(32) void attn_kernel(
    const float* __restrict__ xyz, const int* __restrict__ knn,
    const float* __restrict__ q_all, const float* __restrict__ k_all,
    const float* __restrict__ v_all,
    const float* __restrict__ Wp1, const float* __restrict__ bp1,
    const float* __restrict__ Wp2, const float* __restrict__ bp2,
    _Float16* __restrict__ attn_feat, float* __restrict__ attn_out)
{
  const int point = blockIdx.x;           // b*N + n
  const int b = point / N_;
  const int n = point - b * N_;
  const int lane = threadIdx.x;
  const int hi = lane >> 4, l16 = lane & 15;

  __shared__ int      sidx[KNN_];
  __shared__ float    srel[KNN_][4];
  __shared__ _Float16 sh1[KNN_][HD_];     // relu(rel @ Wp1 + bp1), f16
  __shared__ _Float16 sW2t[HD_][HD_];     // Wp2 transposed (n,k), f16
  __shared__ float    spos[KNN_][HD_];
  __shared__ float    sq[DOUT_];
  __shared__ float    sscore[KNN_][H_];
  __shared__ float    sattn[KNN_][H_];

  if (lane < KNN_) {
    const int id = knn[(size_t)point * KNN_ + lane];
    sidx[lane] = id;
    const float cx = xyz[(size_t)point*3+0];
    const float cy = xyz[(size_t)point*3+1];
    const float cz = xyz[(size_t)point*3+2];
    const float* nx = &xyz[((size_t)b*N_ + id)*3];
    srel[lane][0] = nx[0]-cx; srel[lane][1] = nx[1]-cy;
    srel[lane][2] = nx[2]-cz; srel[lane][3] = 0.f;
  }
#pragma unroll
  for (int t = 0; t < DOUT_/32; ++t)
    sq[lane + 32*t] = q_all[(size_t)point*DOUT_ + lane + 32*t];
  for (int t = lane; t < HD_*HD_; t += 32) {
    const int kk = t >> 6, nn = t & 63;   // Wp2 (64x64) row-major: t = kk*64+nn
    sW2t[nn][kk] = (_Float16)Wp2[t];
  }
  __syncthreads();

  // pos MLP layer 1 (inner dim 3): 16*64 outputs, VALU
  for (int t = lane; t < KNN_*HD_; t += 32) {
    const int kk = t >> 6, j = t & 63;
    float h = bp1[j] + srel[kk][0]*Wp1[j] + srel[kk][1]*Wp1[64+j] + srel[kk][2]*Wp1[128+j];
    sh1[kk][j] = (_Float16)fmaxf(h, 0.f);
  }
  __syncthreads();

  // layer 2 via WMMA: pos(16x64) = sh1(16x64) @ Wp2(64x64) + bp2
  const v8f vzero = {0.f,0.f,0.f,0.f,0.f,0.f,0.f,0.f};
  v8f acc[4] = {vzero, vzero, vzero, vzero};
#pragma unroll
  for (int ks = 0; ks < 2; ++ks) {
    FragH a;
#pragma unroll
    for (int j = 0; j < 8; ++j) {
      const int kk = ks*32 + 16*hi + 2*j;
      a.u[j] = *(const uint32_t*)&sh1[l16][kk];
    }
#pragma unroll
    for (int nt = 0; nt < 4; ++nt) {
      FragH bf;
#pragma unroll
      for (int j = 0; j < 8; ++j) {
        const int kk = ks*32 + 16*hi + 2*j;
        bf.u[j] = *(const uint32_t*)&sW2t[nt*16 + l16][kk];
      }
      acc[nt] = __builtin_amdgcn_wmma_f32_16x16x32_f16(
          false, a.v, false, bf.v, (short)0, acc[nt], false, false);
    }
  }
#pragma unroll
  for (int nt = 0; nt < 4; ++nt) {
    const int ncol = nt*16 + l16;
    const float bb = bp2[ncol];
#pragma unroll
    for (int r = 0; r < 8; ++r)
      spos[r + 8*hi][ncol] = acc[nt][r] + bb;
  }
  __syncthreads();

  // scores: 64 (k,h) pairs, 2 per lane; float4 gathers of k rows
#pragma unroll
  for (int p = 0; p < 2; ++p) {
    const int kh = lane*2 + p;
    const int kk = kh >> 2, h = kh & 3;
    const float4* kr4 = (const float4*)&k_all[((size_t)b*N_ + sidx[kk])*DOUT_ + h*HD_];
    const float* qh = &sq[h*HD_];
    float s = 0.f;
#pragma unroll
    for (int j4 = 0; j4 < HD_/4; ++j4) {
      const float4 kv = kr4[j4];
      const int j = j4*4;
      s += qh[j+0] * (kv.x + spos[kk][j+0]);
      s += qh[j+1] * (kv.y + spos[kk][j+1]);
      s += qh[j+2] * (kv.z + spos[kk][j+2]);
      s += qh[j+3] * (kv.w + spos[kk][j+3]);
    }
    sscore[kk][h] = s * 0.125f;           // 1/sqrt(64)
  }
  __syncthreads();

  // softmax over K per head; also emit attn_out (B,H,N,K)
  if (lane < H_) {
    float mx = -3.0e38f;
#pragma unroll
    for (int kk = 0; kk < KNN_; ++kk) mx = fmaxf(mx, sscore[kk][lane]);
    float sum = 0.f;
#pragma unroll
    for (int kk = 0; kk < KNN_; ++kk) {
      const float e = __expf(sscore[kk][lane] - mx);
      sattn[kk][lane] = e; sum += e;
    }
    const float inv = 1.f / sum;
#pragma unroll
    for (int kk = 0; kk < KNN_; ++kk) {
      const float a = sattn[kk][lane] * inv;
      sattn[kk][lane] = a;
      attn_out[(((size_t)b*H_ + lane)*N_ + n)*KNN_ + kk] = a;
    }
  }
  __syncthreads();

  // out[h][j] = sum_k attn * (v + pos); 8 outputs per lane, float4 v reads
  const int h  = lane >> 3;
  const int j0 = (lane & 7) * 8;
  float o[8] = {0.f,0.f,0.f,0.f,0.f,0.f,0.f,0.f};
  for (int kk = 0; kk < KNN_; ++kk) {
    const float a = sattn[kk][h];
    const float4* vr4 = (const float4*)&v_all[((size_t)b*N_ + sidx[kk])*DOUT_ + h*HD_ + j0];
    const float4 v0 = vr4[0], v1 = vr4[1];
    o[0] += a * (v0.x + spos[kk][j0+0]); o[1] += a * (v0.y + spos[kk][j0+1]);
    o[2] += a * (v0.z + spos[kk][j0+2]); o[3] += a * (v0.w + spos[kk][j0+3]);
    o[4] += a * (v1.x + spos[kk][j0+4]); o[5] += a * (v1.y + spos[kk][j0+5]);
    o[6] += a * (v1.z + spos[kk][j0+6]); o[7] += a * (v1.w + spos[kk][j0+7]);
  }
#pragma unroll
  for (int j = 0; j < 8; ++j)
    attn_feat[(size_t)point*DOUT_ + h*HD_ + j0 + j] = (_Float16)o[j];
}

// ---------------------------------------------------------------------------
// LayerNorm (+optional residual add), f32 or f16 output; blockDim == D
// ---------------------------------------------------------------------------
__global__ void ln_res_kernel(const float* __restrict__ x, const float* __restrict__ g,
                              const float* __restrict__ bb, const float* __restrict__ res,
                              float* __restrict__ y, _Float16* __restrict__ y16, int D)
{
  __shared__ float red[256];
  const int row = blockIdx.x;
  const int t = threadIdx.x;
  const float v = x[(size_t)row*D + t];
  red[t] = v; __syncthreads();
  for (int s = D >> 1; s > 0; s >>= 1) { if (t < s) red[t] += red[t+s]; __syncthreads(); }
  const float mu = red[0] / (float)D;
  __syncthreads();
  const float dv = v - mu;
  red[t] = dv * dv; __syncthreads();
  for (int s = D >> 1; s > 0; s >>= 1) { if (t < s) red[t] += red[t+s]; __syncthreads(); }
  const float var = red[0] / (float)D;
  const float rs = rsqrtf(var + 1e-5f);
  float o = dv * rs * g[t] + bb[t];
  if (res) o += res[(size_t)row*D + t];
  if (y16) y16[(size_t)row*D + t] = (_Float16)o;
  else     y  [(size_t)row*D + t] = o;
}

// ---------------------------------------------------------------------------
extern "C" void kernel_launch(void* const* d_in, const int* in_sizes, int n_in,
                              void* d_out, int out_size, void* d_ws, size_t ws_size,
                              hipStream_t stream)
{
  const float* xyz  = (const float*)d_in[0];
  const float* feat = (const float*)d_in[1];
  const float* Wq  = (const float*)d_in[2];  const float* bq  = (const float*)d_in[3];
  const float* Wk  = (const float*)d_in[4];  const float* bk  = (const float*)d_in[5];
  const float* Wv  = (const float*)d_in[6];  const float* bv  = (const float*)d_in[7];
  const float* Wp1 = (const float*)d_in[8];  const float* bp1 = (const float*)d_in[9];
  const float* Wp2 = (const float*)d_in[10]; const float* bp2 = (const float*)d_in[11];
  const float* Wf1 = (const float*)d_in[12]; const float* bf1 = (const float*)d_in[13];
  const float* Wf2 = (const float*)d_in[14]; const float* bf2 = (const float*)d_in[15];
  const float* g1  = (const float*)d_in[16]; const float* b1  = (const float*)d_in[17];
  const float* g2  = (const float*)d_in[18]; const float* b2  = (const float*)d_in[19];

  float* out_main = (float*)d_out;                       // (B,N,256)
  float* attn_out = out_main + (size_t)M_ * DOUT_;       // (B,H,N,K)

  char* ws = (char*)d_ws;
  const size_t SZ_IDX  = (size_t)M_ * KNN_  * sizeof(int);       // 2 MiB
  const size_t SZ_BIG  = (size_t)M_ * DOUT_ * sizeof(float);     // 33.5 MiB
  const size_t SZ_H128 = (size_t)M_ * DIM_  * sizeof(_Float16);  // 8.4 MiB

  int*      knn    = (int*)ws;
  float*    q_all  = (float*)(ws + SZ_IDX);
  float*    k_all  = (float*)(ws + SZ_IDX + 1*SZ_BIG);
  float*    v_all  = (float*)(ws + SZ_IDX + 2*SZ_BIG);
  float*    res    = (float*)(ws + SZ_IDX + 3*SZ_BIG);
  _Float16* feat_h = (_Float16*)(ws + SZ_IDX + 4*SZ_BIG);
  _Float16* ln1_h  = (_Float16*)(ws + SZ_IDX + 4*SZ_BIG + SZ_H128);
  _Float16* attn_h = (_Float16*)(ws + SZ_IDX + 4*SZ_BIG + 2*SZ_H128);
  _Float16* ffn1_h = feat_h;     // feat_h+ln1_h (16.8 MiB) dead after projections
  float*    fc     = k_all;      // k_all dead after attention

  // 1) KNN indices (sorted ascending distance, matches top_k ordering)
  knn_kernel<<<dim3(N_/QT, B_), QT, 0, stream>>>(xyz, knn);

  // 2) f16 activations: features copy + LN(features) for residual branch
  cvt_f32_f16_kernel<<<(M_*DIM_/4)/256, 256, 0, stream>>>(feat, feat_h);
  ln_res_kernel<<<M_, DIM_, 0, stream>>>(feat, g1, b1, nullptr, nullptr, ln1_h, DIM_);

  // 3) projections (project-then-gather; linear => identical to reference)
  dim3 ggrid(M_/BM, DOUT_/BN);
  gemm_wmma_kernel<<<ggrid, 256, 0, stream>>>(feat_h, Wq, bq, q_all, nullptr, M_, DIM_, DOUT_, 0);
  gemm_wmma_kernel<<<ggrid, 256, 0, stream>>>(feat_h, Wk, bk, k_all, nullptr, M_, DIM_, DOUT_, 0);
  gemm_wmma_kernel<<<ggrid, 256, 0, stream>>>(feat_h, Wv, bv, v_all, nullptr, M_, DIM_, DOUT_, 0);
  gemm_wmma_kernel<<<ggrid, 256, 0, stream>>>(ln1_h,  Wq, bq, res,   nullptr, M_, DIM_, DOUT_, 0);

  // 4) neighborhood attention (pos-MLP layer2 via WMMA), emits attn_out too
  attn_kernel<<<M_, 32, 0, stream>>>(xyz, knn, q_all, k_all, v_all,
                                     Wp1, bp1, Wp2, bp2, attn_h, attn_out);

  // 5) FFN (first GEMM emits f16 directly for the second)
  gemm_wmma_kernel<<<ggrid, 256, 0, stream>>>(attn_h, Wf1, bf1, nullptr, ffn1_h, M_, DOUT_, DOUT_, 1);
  gemm_wmma_kernel<<<ggrid, 256, 0, stream>>>(ffn1_h, Wf2, bf2, fc,      nullptr, M_, DOUT_, DOUT_, 0);

  // 6) final LN + residual
  ln_res_kernel<<<M_, DOUT_, 0, stream>>>(fc, g2, b2, res, out_main, nullptr, DOUT_);

  (void)in_sizes; (void)n_in; (void)out_size; (void)ws_size;
}
